// MultiHeadAttention_44032004719197
// MI455X (gfx1250) — compile-verified, matrix operand order checked by reading, not run on hardware
//
#include <hip/hip_runtime.h>
#include <hip/hip_bf16.h>

// ---------------- problem constants ----------------
#define BB 4
#define TB 2048
#define CC 1024
#define HH 16
#define DD 64
#define NQKV (3 * CC)

// ---------------- WMMA types -----------------------
typedef __attribute__((ext_vector_type(16))) __bf16 v16bf;
typedef __attribute__((ext_vector_type(8)))  float  v8f;
typedef __attribute__((ext_vector_type(4)))  unsigned int v4u;

union FragBF { v16bf v; v4u q[2]; };

static __device__ __forceinline__ v8f wmma_bf16(const FragBF& a, const FragBF& b, v8f c) {
  return __builtin_amdgcn_wmma_f32_16x16x32_bf16(false, a.v, false, b.v, (short)0, c, false, false);
}

// 16-bit A/B fragment layout (ISA 7.12.2): lane = (half<<4)|m, element e maps to
// k = e + 8*half + (e>=8 ? 8 : 0)  -> two contiguous 8-element (16B) groups.
static __device__ __forceinline__ void load_frag_row(FragBF& f, const __bf16* rowptr, int half) {
  f.q[0] = *(const v4u*)(rowptr + 8 * half);
  f.q[1] = *(const v4u*)(rowptr + 16 + 8 * half);
}

// Async global->LDS copy of 16 bytes (8 bf16) per lane. ASYNCcnt-tracked,
// bypasses VGPRs (CDNA5 GLOBAL_LOAD_ASYNC_TO_LDS_B128, GV addressing mode).
static __device__ __forceinline__ void async_cp16(const __bf16* gptr, __bf16* lptr) {
  unsigned lds  = (unsigned)(size_t)lptr;                  // low 32b = LDS offset
  unsigned long long ga = (unsigned long long)(size_t)gptr;
  asm volatile("global_load_async_to_lds_b128 %0, %1, off"
               :: "v"(lds), "v"(ga) : "memory");
}
static __device__ __forceinline__ void wait_async0() {
  asm volatile("s_wait_asynccnt 0x0" ::: "memory");
}

// ---------------- fp32 -> bf16 convert --------------
__global__ __launch_bounds__(256)
void f32_to_bf16_kernel(const float* __restrict__ src, __bf16* __restrict__ dst, int n) {
  int i = blockIdx.x * blockDim.x + threadIdx.x;
  int stride = gridDim.x * blockDim.x;
  for (; i < n; i += stride) dst[i] = (__bf16)src[i];
}

// ---------------- fp32 (K,N) -> bf16 (N,K) weight transpose -----------------
// One-time cost; removes every in-loop transpose from the GEMMs.
__global__ __launch_bounds__(256)
void transpose_w_bf16(const float* __restrict__ src, __bf16* __restrict__ dst,
                      int K, int N) {
  __shared__ float tile[32][33];
  const int n0 = blockIdx.x * 32, k0 = blockIdx.y * 32;
  const int tx = threadIdx.x & 31, ty = threadIdx.x >> 5;  // 8 rows of 32
#pragma unroll
  for (int i = 0; i < 32; i += 8)
    tile[ty + i][tx] = src[(size_t)(k0 + ty + i) * N + n0 + tx];
  __syncthreads();
#pragma unroll
  for (int i = 0; i < 32; i += 8)
    dst[(size_t)(n0 + ty + i) * K + k0 + tx] = (__bf16)tile[tx][ty + i];
}

// ---------------- pipelined WMMA GEMM -------------------
// C[M,N] = A[M,K] * BwT[N,K]^T + bias[N]   (BwT is the pre-transposed weight)
// MODE 0: scatter bf16 into Q/K (B,H,T,D) and V (B,H,D,T); MODE 1: fp32 outF.
template<int MODE>
__global__ __launch_bounds__(256)
void gemm_bf16_wmma(const __bf16* __restrict__ A, const __bf16* __restrict__ BwT,
                    const float* __restrict__ bias,
                    __bf16* __restrict__ outQ, __bf16* __restrict__ outK,
                    __bf16* __restrict__ outVt, float* __restrict__ outF,
                    int M, int N, int K)
{
  __shared__ __align__(16) __bf16 As[2][128 * 32];   // [m][k]
  __shared__ __align__(16) __bf16 Bs[2][128 * 32];   // [n][k]

  const int tid  = threadIdx.x;
  const int lane = tid & 31;
  const int wave = tid >> 5;
  const int wm   = wave & 3;   // 4 waves along M (32 rows each)
  const int wn   = wave >> 2;  // 2 waves along N (64 cols each)
  const int half = lane >> 4;
  const int l16  = lane & 15;
  const int row0 = blockIdx.y * 128;
  const int col0 = blockIdx.x * 128;
  (void)M;

  // staging map: thread covers 16 contiguous bf16 (2 x async b128) per matrix
  const int sr = tid >> 1;            // 0..127 : A row / BwT row
  const int kb = (tid & 1) * 16;      // 0 or 16

  auto stage = [&](int buf, int k0) {
    const __bf16* ga = A   + (size_t)(row0 + sr) * K + k0 + kb;
    async_cp16(ga,     &As[buf][sr * 32 + kb]);
    async_cp16(ga + 8, &As[buf][sr * 32 + kb + 8]);
    const __bf16* gb = BwT + (size_t)(col0 + sr) * K + k0 + kb;
    async_cp16(gb,     &Bs[buf][sr * 32 + kb]);
    async_cp16(gb + 8, &Bs[buf][sr * 32 + kb + 8]);
  };

  v8f acc[2][4];
#pragma unroll
  for (int mi = 0; mi < 2; ++mi)
#pragma unroll
    for (int nj = 0; nj < 4; ++nj) acc[mi][nj] = 0.f;

  const int nsteps = K / 32;
  stage(0, 0);
  for (int s = 0; s < nsteps; ++s) {
    const int cur = s & 1;
    wait_async0();
    __syncthreads();                       // tile[cur] resident for all waves
    if (s + 1 < nsteps) stage(1 - cur, (s + 1) * 32);  // overlap next copy

    FragBF af[2], bf2[4];
#pragma unroll
    for (int mi = 0; mi < 2; ++mi)
      load_frag_row(af[mi], &As[cur][(wm * 32 + mi * 16 + l16) * 32], half);
#pragma unroll
    for (int nj = 0; nj < 4; ++nj)
      load_frag_row(bf2[nj], &Bs[cur][(wn * 64 + nj * 16 + l16) * 32], half);

#pragma unroll
    for (int mi = 0; mi < 2; ++mi)
#pragma unroll
      for (int nj = 0; nj < 4; ++nj)
        acc[mi][nj] = wmma_bf16(af[mi], bf2[nj], acc[mi][nj]);

    __syncthreads();                       // reads of tile[cur] done
  }

  // epilogue: C/D layout -> lane holds rows m = r + 8*half, col n = l16
#pragma unroll
  for (int mi = 0; mi < 2; ++mi) {
#pragma unroll
    for (int nj = 0; nj < 4; ++nj) {
#pragma unroll
      for (int r = 0; r < 8; ++r) {
        const int gm = row0 + wm * 32 + mi * 16 + r + 8 * half;
        const int gn = col0 + wn * 64 + nj * 16 + l16;
        const float v = acc[mi][nj][r] + bias[gn];
        if (MODE == 0) {
          const int bidx = gm / TB, t = gm % TB;
          const int sel = gn / CC, c = gn % CC;
          const int h = c / DD, d = c % DD;
          const size_t bh = (size_t)bidx * HH + h;
          if (sel == 2) {
            outVt[(bh * DD + d) * TB + t] = (__bf16)v;      // V^T: (B,H,D,T)
          } else {
            __bf16* dst = (sel == 0) ? outQ : outK;          // (B,H,T,D)
            dst[(bh * TB + t) * DD + d] = (__bf16)v;
          }
        } else {
          outF[(size_t)gm * N + gn] = v;
        }
      }
    }
  }
}

// ---------------- flash attention -------------------
// grid: (T/64, B*H), 128 threads (4 waves, 16 query rows each).
// K/V^T chunks (32 keys) double-buffered via async LDS copies.
__global__ __launch_bounds__(128)
void attn_fwd_wmma(const __bf16* __restrict__ Q, const __bf16* __restrict__ K,
                   const __bf16* __restrict__ Vt, const int* __restrict__ mask,
                   __bf16* __restrict__ O)
{
  __shared__ __align__(16) __bf16 Ks[2][32 * 64];     // [key][d]
  __shared__ __align__(16) __bf16 VsT[2][64 * 32];    // [d][key]
  __shared__ __align__(16) __bf16 Ps[4][16 * 32];     // per-wave P tile [m][key]

  const int tid  = threadIdx.x;
  const int lane = tid & 31;
  const int wave = tid >> 5;
  const int half = lane >> 4;
  const int l16  = lane & 15;
  const int bh   = blockIdx.y;
  const int b    = bh >> 4;   // / HH
  const int h    = bh & 15;
  const int t0q  = blockIdx.x * 64 + wave * 16;

  const __bf16* Qb  = Q  + (size_t)bh * TB * DD;
  const __bf16* Kb  = K  + (size_t)bh * TB * DD;
  const __bf16* Vtb = Vt + (size_t)bh * DD * TB;

  // staging map (128 threads)
  const int kr  = tid >> 2;            // 0..31  key row
  const int kcb = (tid & 3) * 16;      // d offset 0/16/32/48
  const int vd  = tid >> 1;            // 0..63  d row of V^T
  const int vkb = (tid & 1) * 16;      // key offset 0/16

  auto stage = [&](int buf, int t0k) {
    const __bf16* kg = Kb + (size_t)(t0k + kr) * DD + kcb;
    async_cp16(kg,     &Ks[buf][kr * 64 + kcb]);
    async_cp16(kg + 8, &Ks[buf][kr * 64 + kcb + 8]);
    const __bf16* vg = Vtb + (size_t)vd * TB + t0k + vkb;
    async_cp16(vg,     &VsT[buf][vd * 32 + vkb]);
    async_cp16(vg + 8, &VsT[buf][vd * 32 + vkb + 8]);
  };

  // Q tile 16x64 -> two 16x32 A fragments, kept in registers
  FragBF qf[2];
  {
    const __bf16* qrow = Qb + (size_t)(t0q + l16) * DD;
    load_frag_row(qf[0], qrow, half);
    load_frag_row(qf[1], qrow + 32, half);
  }

  v8f oacc[4];
#pragma unroll
  for (int j = 0; j < 4; ++j) oacc[j] = 0.f;
  float rmax[8], rsum[8];
#pragma unroll
  for (int i = 0; i < 8; ++i) { rmax[i] = -1e30f; rsum[i] = 0.f; }

  stage(0, 0);
  for (int t0k = 0; t0k < TB; t0k += 32) {
    const int cur = (t0k >> 5) & 1;
    wait_async0();
    __syncthreads();                         // chunk[cur] resident
    if (t0k + 32 < TB) stage(1 - cur, t0k + 32);   // overlap next copy

    // S = Q @ K^T for 32 keys (two 16x16 tiles), f32 accum
    v8f s[2];
#pragma unroll
    for (int j = 0; j < 2; ++j) {
      FragBF kf0, kf1;
      const __bf16* krow = &Ks[cur][(j * 16 + l16) * 64];
      load_frag_row(kf0, krow, half);
      load_frag_row(kf1, krow + 32, half);
      v8f z = 0.f;
      s[j] = wmma_bf16(qf[0], kf0, z);
      s[j] = wmma_bf16(qf[1], kf1, s[j]);
    }

    // scale + key-padding mask (lane's column n = l16 -> key t0k + 16j + l16)
    const int mrow = b * TB + t0k + l16;
    const int mk0 = mask[mrow];
    const int mk1 = mask[mrow + 16];
#pragma unroll
    for (int i = 0; i < 8; ++i) {
      s[0][i] = mk0 ? s[0][i] * 0.125f : -1e30f;
      s[1][i] = mk1 ? s[1][i] * 0.125f : -1e30f;
    }

    // online softmax; one row lives across 16 lanes of a half-group
    float p0[8], p1[8];
#pragma unroll
    for (int i = 0; i < 8; ++i) {
      float mx = fmaxf(s[0][i], s[1][i]);
      mx = fmaxf(mx, __shfl_xor(mx, 1, 32));
      mx = fmaxf(mx, __shfl_xor(mx, 2, 32));
      mx = fmaxf(mx, __shfl_xor(mx, 4, 32));
      mx = fmaxf(mx, __shfl_xor(mx, 8, 32));
      const float newmax = fmaxf(rmax[i], mx);
      const float alpha  = __expf(rmax[i] - newmax);
      rmax[i] = newmax;
      p0[i] = (s[0][i] <= -1e29f) ? 0.f : __expf(s[0][i] - newmax);
      p1[i] = (s[1][i] <= -1e29f) ? 0.f : __expf(s[1][i] - newmax);
      float ls = p0[i] + p1[i];
      ls += __shfl_xor(ls, 1, 32);
      ls += __shfl_xor(ls, 2, 32);
      ls += __shfl_xor(ls, 4, 32);
      ls += __shfl_xor(ls, 8, 32);
      rsum[i] = rsum[i] * alpha + ls;
#pragma unroll
      for (int j4 = 0; j4 < 4; ++j4) oacc[j4][i] *= alpha;
    }

    // re-layout P (C/D layout) -> A layout via per-wave LDS tile
    __bf16* pw = &Ps[wave][0];
#pragma unroll
    for (int i = 0; i < 8; ++i) {
      pw[(i + 8 * half) * 32 + l16]      = (__bf16)p0[i];
      pw[(i + 8 * half) * 32 + 16 + l16] = (__bf16)p1[i];
    }
    FragBF pa;
    load_frag_row(pa, &pw[l16 * 32], half);  // LDS in-order within wave

    // O += P @ V   (V^T resident: B fragment rows are contiguous)
#pragma unroll
    for (int j4 = 0; j4 < 4; ++j4) {
      FragBF vf;
      load_frag_row(vf, &VsT[cur][(j4 * 16 + l16) * 32], half);
      oacc[j4] = wmma_bf16(pa, vf, oacc[j4]);
    }
    __syncthreads();                         // reads of chunk[cur] done
  }

  // finalize: O /= rowsum, store bf16 in (B,T,C) for the out-projection GEMM
#pragma unroll
  for (int i = 0; i < 8; ++i) {
    const float inv = 1.0f / rsum[i];
    const int t = t0q + i + 8 * half;
#pragma unroll
    for (int j4 = 0; j4 < 4; ++j4) {
      O[((size_t)(b * TB + t)) * CC + h * DD + j4 * 16 + l16] =
          (__bf16)(oacc[j4][i] * inv);
    }
  }
}

// ---------------- launcher --------------------------
extern "C" void kernel_launch(void* const* d_in, const int* in_sizes, int n_in,
                              void* d_out, int out_size, void* d_ws, size_t ws_size,
                              hipStream_t stream)
{
  const float* input  = (const float*)d_in[0];
  const int*   amask  = (const int*)d_in[1];
  const float* W_qkv  = (const float*)d_in[2];
  const float* b_qkv  = (const float*)d_in[3];
  const float* W_proj = (const float*)d_in[4];
  const float* b_proj = (const float*)d_in[5];
  float* out = (float*)d_out;

  char* ws = (char*)d_ws;
  size_t off = 0;
  auto alloc = [&](size_t bytes) -> char* {
    char* p = ws + off;
    off += (bytes + 255) & ~(size_t)255;
    return p;
  };

  const size_t nX  = (size_t)BB * TB * CC;   // 8,388,608
  const size_t nWq = (size_t)CC * NQKV;      // 3,145,728
  const size_t nWp = (size_t)CC * CC;        // 1,048,576

  __bf16* Xbf  = (__bf16*)alloc(nX  * 2);
  __bf16* WqT  = (__bf16*)alloc(nWq * 2);    // (3072, 1024) = W_qkv^T, bf16
  __bf16* WpT  = (__bf16*)alloc(nWp * 2);    // (1024, 1024) = W_proj^T, bf16
  __bf16* Qbf  = (__bf16*)alloc(nX * 2);     // (B,H,T,D)
  __bf16* Kbf  = (__bf16*)alloc(nX * 2);     // (B,H,T,D)
  __bf16* Vtb  = (__bf16*)alloc(nX * 2);     // (B,H,D,T)
  __bf16* Obf  = (__bf16*)alloc(nX * 2);     // (B*T, C)

  // 1) fp32 -> bf16 conversion of activations; one-time weight transposes
  f32_to_bf16_kernel<<<dim3(4096), dim3(256), 0, stream>>>(input, Xbf, (int)nX);
  transpose_w_bf16<<<dim3(NQKV / 32, CC / 32), dim3(256), 0, stream>>>(
      W_qkv, WqT, CC, NQKV);
  transpose_w_bf16<<<dim3(CC / 32, CC / 32), dim3(256), 0, stream>>>(
      W_proj, WpT, CC, CC);

  // 2) QKV projection: (8192 x 3072 x 1024); epilogue scatters Q,K,(V^T)
  gemm_bf16_wmma<0><<<dim3(NQKV / 128, (BB * TB) / 128), dim3(256), 0, stream>>>(
      Xbf, WqT, b_qkv, Qbf, Kbf, Vtb, nullptr, BB * TB, NQKV, CC);

  // 3) flash attention per (b,h), 64 query rows per block
  attn_fwd_wmma<<<dim3(TB / 64, BB * HH), dim3(128), 0, stream>>>(
      Qbf, Kbf, Vtb, amask, Obf);

  // 4) output projection: (8192 x 1024 x 1024), fp32 epilogue + bias -> d_out
  gemm_bf16_wmma<1><<<dim3(CC / 128, (BB * TB) / 128), dim3(256), 0, stream>>>(
      Obf, WpT, b_proj, nullptr, nullptr, nullptr, out, BB * TB, CC, CC);
}
